// LoopMoE_5291399708788
// MI455X (gfx1250) — matrix operand order repaired and checked by reading, not compiled, not run on hardware
//
#include <hip/hip_runtime.h>
#include <hip/hip_bf16.h>
#include <math.h>

#define HIDDEN   2048
#define INTER    4096
#define NEXP     8
#define TOKENS   8192
#define TM       16     // tokens per block tile
#define TI       128    // inter-dim chunk per iteration (16 cols per wave x 8 waves)
#define KS       32     // K-slice per WMMA step
#define NSLICE   (HIDDEN / KS)
#define NTHREADS 256    // 8 wave32 waves

typedef __bf16 bf16_t;
typedef __bf16    v16bf __attribute__((ext_vector_type(16)));
typedef float     v8f   __attribute__((ext_vector_type(8)));
typedef unsigned  v4u   __attribute__((ext_vector_type(4)));
typedef int       v8i   __attribute__((ext_vector_type(8)));
typedef int       v4i   __attribute__((ext_vector_type(4)));

// ---------------------------------------------------------------------------
// Phase 1: router. One block per token: logits -> softmax -> top-2 -> scatter
// into per-expert compacted lists (counts via atomicAdd).
// ---------------------------------------------------------------------------
__global__ __launch_bounds__(NTHREADS) void moe_router_kernel(
    const float* __restrict__ x, const float* __restrict__ rw,
    int* __restrict__ counts, int* __restrict__ tok_list,
    float* __restrict__ tok_w)
{
    __shared__ float xs[HIDDEN];
    __shared__ float part[NEXP][NTHREADS];

    const int t = blockIdx.x;
    const float* xr = x + (size_t)t * HIDDEN;
    for (int h = threadIdx.x; h < HIDDEN; h += NTHREADS) xs[h] = xr[h];
    __syncthreads();

    float acc[NEXP];
#pragma unroll
    for (int e = 0; e < NEXP; ++e) acc[e] = 0.f;
    for (int h = threadIdx.x; h < HIDDEN; h += NTHREADS) {
        float xv = xs[h];
#pragma unroll
        for (int e = 0; e < NEXP; ++e) acc[e] += xv * rw[e * HIDDEN + h];
    }
#pragma unroll
    for (int e = 0; e < NEXP; ++e) part[e][threadIdx.x] = acc[e];
    __syncthreads();

    for (int s = NTHREADS / 2; s > 0; s >>= 1) {
        if (threadIdx.x < s) {
#pragma unroll
            for (int e = 0; e < NEXP; ++e)
                part[e][threadIdx.x] += part[e][threadIdx.x + s];
        }
        __syncthreads();
    }

    if (threadIdx.x == 0) {
        float p[NEXP];
        float m = part[0][0];
        for (int e = 1; e < NEXP; ++e) m = fmaxf(m, part[e][0]);
        float s = 0.f;
        for (int e = 0; e < NEXP; ++e) { p[e] = __expf(part[e][0] - m); s += p[e]; }
        const float inv = 1.f / s;
        int i0 = 0;
        for (int e = 1; e < NEXP; ++e) if (p[e] > p[i0]) i0 = e;
        int i1 = (i0 == 0) ? 1 : 0;
        for (int e = 0; e < NEXP; ++e) if (e != i0 && p[e] > p[i1]) i1 = e;
        const int sel[2] = {i0, i1};
#pragma unroll
        for (int k = 0; k < 2; ++k) {
            const int e = sel[k];
            const int pos = atomicAdd(&counts[e], 1);
            tok_list[e * TOKENS + pos] = t;
            tok_w[e * TOKENS + pos]    = p[e] * inv;   // renormalize=False: raw prob
        }
    }
}

// ---------------------------------------------------------------------------
// WMMA fragment loaders (wave32; contiguous k order, A and B alike, not the ISA table order).
// A 16x32 bf16: lanes 0-15 -> M=lane, K=0..15; lanes 16-31 -> M=lane-16, K=16..31.
// B 32x16 bf16: lanes 0-15 -> N=lane, K=0..15; lanes 16-31 -> N=lane-16, K=16..31.
// ---------------------------------------------------------------------------
__device__ __forceinline__ v16bf a_frag_lds(const bf16_t* base, int ld, int k0, int lane)
{
    const int m  = lane & 15;
    const int hi = lane >> 4;
    const bf16_t* row = base + m * ld + k0 + hi * 16;
    v16bf a;
#pragma unroll
    for (int v = 0; v < 8; ++v) {
        const int kb = 2 * v;
        a[2 * v]     = row[kb];
        a[2 * v + 1] = row[kb + 1];
    }
    return a;
}

// B fragment from a TDM-staged fp32 LDS tile [rows][KS] (row = chunk-local N).
__device__ __forceinline__ v16bf b_frag_lds_f32(const float* tile, int n0, int lane)
{
    const int n  = n0 + (lane & 15);
    const int hi = lane >> 4;
    const float* p = tile + n * KS + hi * 16;
    v16bf b;
#pragma unroll
    for (int v = 0; v < 8; ++v) {
        float2 f = *(const float2*)(p + 2 * v);
        b[2 * v]     = (bf16_t)f.x;
        b[2 * v + 1] = (bf16_t)f.y;
    }
    return b;
}

// B fragment straight from a row-major [N][K] fp32 weight matrix in global.
__device__ __forceinline__ v16bf b_frag_global(const float* __restrict__ w, int ld,
                                               int n0, int k0, int lane)
{
    const int n  = n0 + (lane & 15);
    const int hi = lane >> 4;
    const float* p = w + (size_t)n * ld + k0 + hi * 16;
    v16bf b;
#pragma unroll
    for (int v = 0; v < 8; ++v) {
        float2 f = *(const float2*)(p + 2 * v);
        b[2 * v]     = (bf16_t)f.x;
        b[2 * v + 1] = (bf16_t)f.y;
    }
    return b;
}

// ---------------------------------------------------------------------------
// Tensor Data Mover: DMA one K-slice of w1 (gate+up halves) into LDS.
// 3D tile: dim0 = KS contiguous K-columns, dim1 = TI rows, dim2 = {gate,up}
// with tensor_dim1_stride = INTER*HIDDEN elements. D# packed per ISA 8.3-8.6.
// ---------------------------------------------------------------------------
__device__ __forceinline__ void tdm_load_w1_slice(const float* gptr, unsigned ldsoff)
{
    const unsigned long long ga = (unsigned long long)(size_t)gptr;
    v4u g0;
    g0[0] = 1u;                                   // count=1, user desc, no gather
    g0[1] = ldsoff;                               // lds_addr (bytes)
    g0[2] = (unsigned)ga;                         // global_addr[31:0]
    g0[3] = (unsigned)(ga >> 32) | (2u << 30);    // global_addr[56:32] | type=2

    v8i g1;
    g1[0] = (2 << 16);                                        // data_size = 4B
    g1[1] = (int)((HIDDEN & 0xFFFF) << 16);                   // tensor_dim0 lo16
    g1[2] = (int)((HIDDEN >> 16) | ((INTER & 0xFFFF) << 16)); // dim0 hi | dim1 lo
    g1[3] = (int)((INTER >> 16) | (KS << 16));                // dim1 hi | tile_dim0
    g1[4] = (int)(TI | (2 << 16));                            // tile_dim1 | tile_dim2=2
    g1[5] = HIDDEN;                                           // dim0_stride[31:0]
    g1[6] = (int)(((INTER * HIDDEN) & 0xFFFF) << 16);         // d0s hi16 | d1s[15:0]
    g1[7] = (int)((INTER * HIDDEN) >> 16);                    // dim1_stride[47:16]

    v4i g2;
    g2[0] = 2;                                    // tensor_dim2 = {gate, up}
    g2[1] = 0; g2[2] = 0; g2[3] = 0;
    v4i g3 = {0, 0, 0, 0};

#if __has_include(<hip/amd_detail/amd_gfx1250_TDM.h>)
    v8i g4 = {0, 0, 0, 0, 0, 0, 0, 0};
    __builtin_amdgcn_tensor_load_to_lds(g0, g1, g2, g3, g4, 0);
#else
    __builtin_amdgcn_tensor_load_to_lds(g0, g1, g2, g3, 0);
#endif
}

// ---------------------------------------------------------------------------
// Phase 2: fused expert FFN over routed token tiles.
// grid = (TOKENS/TM, NEXP). Block: 16 gathered tokens; full H output kept as
// WMMA C-tiles in registers (16 tiles x v8f per wave = 256 cols/wave).
// GEMM1 B-operands arrive via double-buffered TDM; GEMM2 streams w2 directly
// with global_prefetch ahead of use.
// ---------------------------------------------------------------------------
__global__ __launch_bounds__(NTHREADS, 1) void moe_expert_kernel(
    const float* __restrict__ x,
    const float* __restrict__ w1, const float* __restrict__ w2,
    const int* __restrict__ counts, const int* __restrict__ tok_list,
    const float* __restrict__ tok_w, float* __restrict__ out)
{
    extern __shared__ char smem_raw[];
    bf16_t* xb     = (bf16_t*)smem_raw;               // [TM][HIDDEN] tokens (bf16)
    bf16_t* hb     = xb + TM * HIDDEN;                // [TM][TI] SwiGLU chunk (bf16)
    float*  wstage = (float*)(hb + TM * TI);          // [2][2][TI][KS] fp32 w1 slices
    __shared__ int   toks[TM];
    __shared__ float tws[TM];

    const int e    = blockIdx.y;
    const int tile = blockIdx.x;
    const int cnt  = counts[e];
    if (tile * TM >= cnt) return;             // uniform: EXEC stays all-ones

    const float* w1e = w1 + (size_t)e * (2 * INTER) * HIDDEN;
    const float* w2e = w2 + (size_t)e * HIDDEN * INTER;

    if (threadIdx.x < TM) {
        const int idx = tile * TM + threadIdx.x;
        toks[threadIdx.x] = (idx < cnt) ? tok_list[e * TOKENS + idx] : -1;
        tws[threadIdx.x]  = (idx < cnt) ? tok_w[e * TOKENS + idx] : 0.f;
    }
    __syncthreads();

    for (int i = threadIdx.x; i < TM * HIDDEN; i += NTHREADS) {
        const int m = i >> 11;                // HIDDEN == 2048
        const int h = i & (HIDDEN - 1);
        const int tok = toks[m];
        xb[i] = (tok >= 0) ? (bf16_t)x[(size_t)tok * HIDDEN + h] : (bf16_t)0.f;
    }
    __syncthreads();

    const int lane = threadIdx.x & 31;
    const int wv   = threadIdx.x >> 5;        // 0..7
    const int hi   = lane >> 4;

    const unsigned wstage_off  = (unsigned)(size_t)wstage;   // LDS byte offset
    const unsigned slice_bytes = 2u * TI * KS * sizeof(float);

    v8f zacc[16];
#pragma unroll
    for (int nt = 0; nt < 16; ++nt) {
        v8f z = {0.f, 0.f, 0.f, 0.f, 0.f, 0.f, 0.f, 0.f};
        zacc[nt] = z;
    }

    for (int ic = 0; ic < INTER / TI; ++ic) {
        const float* w1slice = w1e + (size_t)(ic * TI) * HIDDEN;

        // ---- GEMM1: gate & up for this wave's 16 chunk columns, K = HIDDEN.
        // B operands DMA'd by the TDM, double buffered, pipelined one ahead.
        if (wv == 0) tdm_load_w1_slice(w1slice, wstage_off);
        v8f cg = {0.f, 0.f, 0.f, 0.f, 0.f, 0.f, 0.f, 0.f};
        v8f cu = {0.f, 0.f, 0.f, 0.f, 0.f, 0.f, 0.f, 0.f};
        for (int ks = 0; ks < NSLICE; ++ks) {
            const int cur = ks & 1;
            if (ks + 1 < NSLICE) {
                if (wv == 0)
                    tdm_load_w1_slice(w1slice + (ks + 1) * KS,
                                      wstage_off + (unsigned)(cur ^ 1) * slice_bytes);
                __builtin_amdgcn_s_wait_tensorcnt(1);   // oldest DMA done (wave 0)
            } else {
                __builtin_amdgcn_s_wait_tensorcnt(0);   // last DMA done
            }
            __syncthreads();                            // publish buf[cur]

            const float* wb = wstage + cur * (2 * TI * KS);
            v16bf a  = a_frag_lds(xb, HIDDEN, ks * KS, lane);
            v16bf bg = b_frag_lds_f32(wb, wv * 16, lane);               // gate (z=0)
            cg = __builtin_amdgcn_wmma_f32_16x16x32_bf16(false, a, false, bg,
                                                         (short)0, cg, false, false);
            v16bf bu = b_frag_lds_f32(wb + TI * KS, wv * 16, lane);     // up (z=1)
            cu = __builtin_amdgcn_wmma_f32_16x16x32_bf16(false, a, false, bu,
                                                         (short)0, cu, false, false);
            __syncthreads();                            // done reading buf[cur]
        }

        // ---- SwiGLU in registers -> bf16 chunk in LDS ----
        {
            const int ncol = wv * 16 + (lane & 15);
#pragma unroll
            for (int r = 0; r < 8; ++r) {
                const int m = r + 8 * hi;     // C layout: M = r + 8*(lane>=16)
                const float g  = cg[r];
                const float u  = cu[r];
                const float sv = g * __builtin_amdgcn_rcpf(1.f + __expf(-g)); // silu
                hb[m * TI + ncol] = (bf16_t)(u * sv);
            }
        }
        __syncthreads();

        // ---- GEMM2: zacc += h_chunk @ w2_chunk^T  (K = TI) ----
        v16bf ah[4];
#pragma unroll
        for (int kk = 0; kk < 4; ++kk) ah[kk] = a_frag_lds(hb, TI, kk * KS, lane);
#pragma unroll
        for (int nt = 0; nt < 16; ++nt) {
            const int n0 = wv * 256 + nt * 16;
            if (ic + 1 < INTER / TI) {        // stream-ahead into L2/WGP$
                const float* pf = w2e + (size_t)(n0 + (lane & 15)) * INTER +
                                  (ic + 1) * TI + hi * 16;
                __builtin_prefetch(pf, 0, 1);
            }
#pragma unroll
            for (int kk = 0; kk < 4; ++kk) {
                v16bf bw = b_frag_global(w2e, INTER, n0, ic * TI + kk * KS, lane);
                zacc[nt] = __builtin_amdgcn_wmma_f32_16x16x32_bf16(false, ah[kk], false, bw,
                                                                   (short)0, zacc[nt],
                                                                   false, false);
            }
        }
        __syncthreads();                      // hb reused next chunk
    }

    // ---- epilogue: scale by routing prob, atomic combine into out ----
#pragma unroll
    for (int nt = 0; nt < 16; ++nt) {
        const int n = wv * 256 + nt * 16 + (lane & 15);
#pragma unroll
        for (int r = 0; r < 8; ++r) {
            const int m = r + 8 * hi;
            const int tok = toks[m];
            if (tok >= 0)
                atomicAdd(out + (size_t)tok * HIDDEN + n, zacc[nt][r] * tws[m]);
        }
    }
}

// ---------------------------------------------------------------------------
extern "C" void kernel_launch(void* const* d_in, const int* in_sizes, int n_in,
                              void* d_out, int out_size, void* d_ws, size_t ws_size,
                              hipStream_t stream)
{
    const float* x  = (const float*)d_in[0];   // [4, 2048, 2048]
    const float* w1 = (const float*)d_in[1];   // [8, 8192, 2048]
    const float* w2 = (const float*)d_in[2];   // [8, 2048, 4096]
    const float* rw = (const float*)d_in[3];   // [8, 2048]
    float* out = (float*)d_out;                // [4, 2048, 2048]

    // workspace layout: counts | tok_list | tok_w   (~0.5 MB)
    int*   counts   = (int*)d_ws;
    int*   tok_list = (int*)((char*)d_ws + 256);
    float* tok_w    = (float*)((char*)d_ws + 256 + (size_t)NEXP * TOKENS * sizeof(int));

    hipMemsetAsync(counts, 0, NEXP * sizeof(int), stream);
    hipMemsetAsync(out, 0, (size_t)TOKENS * HIDDEN * sizeof(float), stream);

    moe_router_kernel<<<TOKENS, NTHREADS, 0, stream>>>(x, rw, counts, tok_list, tok_w);

    const size_t dyn = (size_t)TM * HIDDEN * sizeof(bf16_t)    // xb   64 KB
                     + (size_t)TM * TI * sizeof(bf16_t)        // hb    4 KB
                     + 2u * 2u * TI * KS * sizeof(float);      // wstage 64 KB
    hipFuncSetAttribute((const void*)moe_expert_kernel,
                        hipFuncAttributeMaxDynamicSharedMemorySize, (int)dyn);
    dim3 grid(TOKENS / TM, NEXP);   // expert-major scheduling keeps 96 MB/expert in L2
    moe_expert_kernel<<<grid, NTHREADS, dyn, stream>>>(x, w1, w2, counts,
                                                       tok_list, tok_w, out);
}
